// AbsoluteSinusoidal2DPE_66151086293478
// MI455X (gfx1250) — matrix operand order repaired and checked by reading, not compiled, not run on hardware
//
#include <hip/hip_runtime.h>
#include <hip/hip_bf16.h>

// Problem geometry (fixed by the reference)
#define DDIM   512
#define PPOS   1024      // 32*32 flattened positions
#define NBATCH 16

// Tiling
#define BM 128
#define BN 128
#define BK 32
#define LDT 40           // padded LDS row stride in bf16 elements (80 B)

typedef __attribute__((ext_vector_type(4)))  float   v4f;
typedef __attribute__((ext_vector_type(8)))  float   v8f;
typedef __attribute__((ext_vector_type(4)))  __bf16  v4bf;
typedef __attribute__((ext_vector_type(8)))  __bf16  v8bf;
typedef __attribute__((ext_vector_type(16))) __bf16  v16bf;

// One BK=32 K-slab: load A/B fragments from LDS per the CDNA5 16-bit WMMA
// operand layouts and issue 4x2 v_wmma_f32_16x16x32_bf16.
__device__ __forceinline__ void compute_step(const __bf16* ldsA, const __bf16* ldsB,
                                             v8f acc[4][2],
                                             int wm, int wn, int mloc, int half) {
  v16bf a[4];
#pragma unroll
  for (int im = 0; im < 4; ++im) {
    // A 16x32 bf16: lane m = mloc; K chunks [8h, 8h+8) and [16+8h, 16+8h+8)
    const __bf16* base = ldsA + (wm + im * 16 + mloc) * LDT;
    v8bf lo = *(const v8bf*)(base + 8 * half);
    v8bf hi = *(const v8bf*)(base + 16 + 8 * half);
    a[im] = __builtin_shufflevector(lo, hi, 0,1,2,3,4,5,6,7,8,9,10,11,12,13,14,15);
  }
  v16bf bfrag[2];
#pragma unroll
  for (int jn = 0; jn < 2; ++jn) {
    // B 32x16 bf16: lane col n = mloc; K chunks [8h, 8h+8) and [16+8h, 16+8h+8)
    const __bf16* base = ldsB + (wn + jn * 16 + mloc) * LDT + 8 * half;
    v8bf lo = *(const v8bf*)(base);
    v8bf hi = *(const v8bf*)(base + 16);
    bfrag[jn] = __builtin_shufflevector(lo, hi, 0,1,2,3,4,5,6,7,8,9,10,11,12,13,14,15);
  }
#pragma unroll
  for (int im = 0; im < 4; ++im)
#pragma unroll
    for (int jn = 0; jn < 2; ++jn)
      acc[im][jn] = __builtin_amdgcn_wmma_f32_16x16x32_bf16(
          false, a[im], false, bfrag[jn], (short)0, acc[im][jn], false, false);
}

// 4x v4f global loads at (uniform base) + (precomputed 32-bit lane offset) + d0
__device__ __forceinline__ void issue4(const float* __restrict__ base,
                                       const int* offs, int d0, v4f* dst) {
#pragma unroll
  for (int r = 0; r < 4; ++r)
    dst[r] = *(const v4f*)(base + offs[r] + d0);
}

// Convert staged f32 to bf16 and commit A/B tiles into an LDS buffer.
__device__ __forceinline__ void commitAB(__bf16* lA, __bf16* lB, const int* offL,
                                         const v4f* sa, const v4f* sb) {
#pragma unroll
  for (int r = 0; r < 4; ++r) {
    *(v4bf*)(lA + offL[r]) = __builtin_convertvector(sa[r], v4bf);
    *(v4bf*)(lB + offL[r]) = __builtin_convertvector(sb[r], v4bf);
  }
}

// Segment-2 variant: B tile = K + E (fold the add at commit time, after compute)
__device__ __forceinline__ void commitAB2(__bf16* lA, __bf16* lB, const int* offL,
                                          const v4f* sa, const v4f* sb, const v4f* sb2) {
#pragma unroll
  for (int r = 0; r < 4; ++r) {
    *(v4bf*)(lA + offL[r]) = __builtin_convertvector(sa[r], v4bf);
    *(v4bf*)(lB + offL[r]) = __builtin_convertvector(sb[r] + sb2[r], v4bf);
  }
}

__global__ __launch_bounds__(256)
void pe2d_logits_kernel(const float* __restrict__ q,
                        const float* __restrict__ k,
                        const float* __restrict__ e,
                        float* __restrict__ out) {
  // Double-buffered LDS tiles (40 KB of 320 KB/WGP)
  __shared__ __bf16 ldsA[2][BM * LDT];
  __shared__ __bf16 ldsB[2][BN * LDT];

  const int tid  = threadIdx.x;
  const int lane = tid & 31;
  const int wave = tid >> 5;
  const int wm   = (wave & 1) * 64;    // wave's row offset inside tile (2 x 64)
  const int wn   = (wave >> 1) * 32;   // wave's col offset inside tile (4 x 32)
  const int mloc = lane & 15;
  const int half = lane >> 4;

  const int tileN = blockIdx.x * BN;
  const int tileM = blockIdx.y * BM;
  const int b     = blockIdx.z;

  const float* qb = q + (size_t)b * PPOS * DDIM;
  const float* kb = k + (size_t)b * PPOS * DDIM;

  // Per-thread 32-bit offsets for the cooperative 128x32 tile loads/stores
  int offA[4], offB[4], offL[4];
#pragma unroll
  for (int r = 0; r < 4; ++r) {
    int lin = tid + r * 256;            // 0..1023 (128 rows * 8 vec4/row)
    int row = lin >> 3;                 // 0..127
    int cv  = (lin & 7) << 2;           // 0,4,...,28
    offA[r] = (tileM + row) * DDIM + cv;
    offB[r] = (tileN + row) * DDIM + cv;
    offL[r] = row * LDT + cv;
  }

  v8f acc[4][2];
#pragma unroll
  for (int im = 0; im < 4; ++im)
#pragma unroll
    for (int jn = 0; jn < 2; ++jn)
      acc[im][jn] = (v8f)(0.0f);

  v4f sa[4], sb[4], sb2[4];

  // Prologue: commit segment-1 step 0 (A = Q_b, B = E) into buffer 0.
  issue4(qb, offA, 0, sa);
  issue4(e,  offB, 0, sb);
  commitAB(ldsA[0], ldsB[0], offL, sa, sb);

  // Segment 1, steps 0..14: compute step s, prefetch step s+1.
  for (int s = 0; s < 15; ++s) {
    const int cur = s & 1;
    issue4(qb, offA, (s + 1) * BK, sa);
    issue4(e,  offB, (s + 1) * BK, sb);
    __syncthreads();
    compute_step(ldsA[cur], ldsB[cur], acc, wm, wn, mloc, half);
    commitAB(ldsA[cur ^ 1], ldsB[cur ^ 1], offL, sa, sb);
  }

  // Transition: compute seg1 step 15 (buffer 1); prefetch seg2 step 0 (buffer 0).
  issue4(e,  offA, 0, sa);                 // A = E rows [tileM..)
  issue4(kb, offB, 0, sb);                 // B = K_b rows [tileN..)
  issue4(e,  offB, 0, sb2);                //   + E (added at commit)
  __syncthreads();
  compute_step(ldsA[1], ldsB[1], acc, wm, wn, mloc, half);
  commitAB2(ldsA[0], ldsB[0], offL, sa, sb, sb2);

  // Segment 2, steps 0..14: compute step s, prefetch step s+1.
  for (int s = 0; s < 15; ++s) {
    const int cur = s & 1;
    issue4(e,  offA, (s + 1) * BK, sa);
    issue4(kb, offB, (s + 1) * BK, sb);
    issue4(e,  offB, (s + 1) * BK, sb2);
    __syncthreads();
    compute_step(ldsA[cur], ldsB[cur], acc, wm, wn, mloc, half);
    commitAB2(ldsA[cur ^ 1], ldsB[cur ^ 1], offL, sa, sb, sb2);
  }

  // Epilogue: compute seg2 step 15 (buffer 1).
  __syncthreads();
  compute_step(ldsA[1], ldsB[1], acc, wm, wn, mloc, half);

  // C/D 16x16 f32 layout: VGPR v -> row (v + 8*half), col = mloc.
  // Output is written once and never re-read: stream it (non-temporal).
#pragma unroll
  for (int im = 0; im < 4; ++im)
#pragma unroll
    for (int jn = 0; jn < 2; ++jn) {
      const int col  = tileN + wn + jn * 16 + mloc;
      const int rowb = tileM + wm + im * 16 + half * 8;
      float* p = out + ((size_t)b * PPOS + rowb) * PPOS + col;
#pragma unroll
      for (int v = 0; v < 8; ++v)
        __builtin_nontemporal_store(acc[im][jn][v], p + (size_t)v * PPOS);
    }
}

extern "C" void kernel_launch(void* const* d_in, const int* in_sizes, int n_in,
                              void* d_out, int out_size, void* d_ws, size_t ws_size,
                              hipStream_t stream) {
  const float* q = (const float*)d_in[0];   // [16,32,32,512]
  const float* k = (const float*)d_in[1];   // [16,32,32,512]
  const float* e = (const float*)d_in[2];   // [32,32,512]
  float* out = (float*)d_out;               // [16,32,32,32,32]

  dim3 grid(PPOS / BN, PPOS / BM, NBATCH);
  dim3 block(256);
  pe2d_logits_kernel<<<grid, block, 0, stream>>>(q, k, e, out);
}
